// GraphConvolution_4698694222238
// MI455X (gfx1250) — matrix operand order repaired and checked by reading, not compiled, not run on hardware
//
#include <hip/hip_runtime.h>
#include <stdint.h>

#define N_NODES 50000
#define DIN 512
#define DOUT 128
#define NEDGE 800000
#define NSUP 2

typedef __attribute__((ext_vector_type(16))) __bf16 bf16x16;
typedef __attribute__((ext_vector_type(8)))  __bf16 bf16x8;
typedef __attribute__((ext_vector_type(8)))  float  v8f;
typedef __attribute__((ext_vector_type(4)))  int    v4i;

// --------------------------------------------------------------------------
// CDNA5 async global->LDS copy path (ASYNCcnt-tracked), with sync fallback.
// Builtin signature (from hipcc diagnostic): (v4i AS1*, v4i AS3*, int, int).
// --------------------------------------------------------------------------
#if __has_builtin(__builtin_amdgcn_global_load_async_to_lds_b128)
#define HAVE_ASYNC_LDS 1
typedef __attribute__((address_space(1))) v4i gv4i;
typedef __attribute__((address_space(3))) v4i lv4i;
__device__ __forceinline__ void async_cp16(const void* g, void* l) {
  __builtin_amdgcn_global_load_async_to_lds_b128(
      (gv4i*)(v4i*)g, (lv4i*)(v4i*)l, 0, 0);
}
template <int N>
__device__ __forceinline__ void wait_asynccnt() {
#if __has_builtin(__builtin_amdgcn_s_wait_asynccnt)
  __builtin_amdgcn_s_wait_asynccnt((short)N);
#else
  asm volatile("s_wait_asynccnt %0" :: "i"(N) : "memory");
#endif
}
#else
#define HAVE_ASYNC_LDS 0
#endif

// --------------------------------------------------------------------------
// Exact JAX threefry2x32 for key(42) -> key words (0, 42).
// --------------------------------------------------------------------------
__device__ __forceinline__ void threefry2x32_k42(uint32_t x0, uint32_t x1,
                                                 uint32_t& o0, uint32_t& o1) {
  const uint32_t ks0 = 0u, ks1 = 42u;
  const uint32_t ks2 = 0x1BD11BDAu ^ ks0 ^ ks1;
  x0 += ks0; x1 += ks1;
#define TF_R(r) { x0 += x1; x1 = (x1 << (r)) | (x1 >> (32 - (r))); x1 ^= x0; }
  TF_R(13) TF_R(15) TF_R(26) TF_R(6)   x0 += ks1; x1 += ks2 + 1u;
  TF_R(17) TF_R(29) TF_R(16) TF_R(24)  x0 += ks2; x1 += ks0 + 2u;
  TF_R(13) TF_R(15) TF_R(26) TF_R(6)   x0 += ks0; x1 += ks1 + 3u;
  TF_R(17) TF_R(29) TF_R(16) TF_R(24)  x0 += ks1; x1 += ks2 + 4u;
  TF_R(13) TF_R(15) TF_R(26) TF_R(6)   x0 += ks2; x1 += ks0 + 5u;
#undef TF_R
  o0 = x0; o1 = x1;
}

__device__ __forceinline__ unsigned short f32_to_bf16(float f) {
  uint32_t u = __float_as_uint(f);
  u += 0x7FFFu + ((u >> 16) & 1u);   // round-to-nearest-even
  return (unsigned short)(u >> 16);
}
__device__ __forceinline__ float bf16_to_f32(unsigned short h) {
  return __uint_as_float(((uint32_t)h) << 16);
}

// --------------------------------------------------------------------------
// Kernel 1: dropout(x) (exact JAX mask, scale 2x) + cast to bf16.
// --------------------------------------------------------------------------
__global__ __launch_bounds__(256)
void dropout_cast(const float* __restrict__ x, unsigned short* __restrict__ xd) {
  const uint32_t half = (uint32_t)(N_NODES * DIN) / 2u;   // 12,800,000
  uint32_t i0 = (blockIdx.x * 256u + threadIdx.x) * 4u;
  if (i0 >= half) return;
  float4 a = *(const float4*)(x + i0);
  float4 b = *(const float4*)(x + half + i0);
  float av[4] = {a.x, a.y, a.z, a.w};
  float bv[4] = {b.x, b.y, b.z, b.w};
  unsigned short ra[4], rb[4];
#pragma unroll
  for (int j = 0; j < 4; ++j) {
    uint32_t o0, o1;
    threefry2x32_k42(i0 + j, half + i0 + j, o0, o1);
    ra[j] = f32_to_bf16((o0 >> 31) ? 0.0f : av[j] * 2.0f);  // keep iff bits<2^31
    rb[j] = f32_to_bf16((o1 >> 31) ? 0.0f : bv[j] * 2.0f);
  }
  *(ushort4*)(xd + i0)        = make_ushort4(ra[0], ra[1], ra[2], ra[3]);
  *(ushort4*)(xd + half + i0) = make_ushort4(rb[0], rb[1], rb[2], rb[3]);
}

// --------------------------------------------------------------------------
// Kernel 1b: one-shot W (f32 [S][K][N]) -> Wt (bf16 [S][N][K]) transpose.
// --------------------------------------------------------------------------
__global__ __launch_bounds__(256)
void wt_cast(const float* __restrict__ W, unsigned short* __restrict__ Wt) {
  int idx = blockIdx.x * 256 + threadIdx.x;   // 131072 total
  int s = idx >> 16;
  int r = idx & 0xFFFF;
  int k = r >> 7;
  int n = r & 127;
  Wt[((size_t)s * DOUT + n) * DIN + k] = f32_to_bf16(W[idx]);
}

// --------------------------------------------------------------------------
// Kernel 2: pre_sup[s] = xd @ W[s]  (bf16 WMMA, fp32 acc, bf16 out).
// 256 threads = 8 waves -> 128 rows; wave owns 16x128 strip (8 v8f accs).
// Double-buffered LDS tiles filled by async global->LDS b128 copies.
// --------------------------------------------------------------------------
#define BM  128
#define KT  32
#define LDA 40   // padded LDS stride (bf16): 80B rows -> conflict-free b128

__global__ __launch_bounds__(256)
void gemm_wmma(const unsigned short* __restrict__ xd,
               const unsigned short* __restrict__ Wt,
               unsigned short* __restrict__ pre) {
  __shared__ unsigned short Alds[2][BM * LDA];
  __shared__ unsigned short Blds[2][DOUT * LDA];

  const int tid  = threadIdx.x;
  const int wave = tid >> 5;
  const int lane = tid & 31;
  const int m16  = lane & 15;
  const int hi   = lane >> 4;
  const int s    = blockIdx.y;
  const int gm0  = blockIdx.x * BM;
  const unsigned short* WtS = Wt + (size_t)s * DOUT * DIN;

  v8f acc[8];
#pragma unroll
  for (int i = 0; i < 8; ++i)
#pragma unroll
    for (int j = 0; j < 8; ++j) acc[i][j] = 0.0f;

#if HAVE_ASYNC_LDS
  // Each thread copies 2x16B of A and 2x16B of B per tile (uniform count of 4
  // async ops per lane -> per-wave ASYNCcnt bookkeeping stays uniform; OOB A
  // rows are clamped, their C rows are discarded at the store).
  auto issue = [&](int buf, int kt) {
#pragma unroll
    for (int j = 0; j < 2; ++j) {
      int lin = tid + j * 256;           // 512 chunks x 16B = 8KB tile
      int row = lin >> 2;
      int c8  = (lin & 3) << 3;
      int grow = gm0 + row;
      if (grow >= N_NODES) grow = N_NODES - 1;
      async_cp16(xd  + (size_t)grow * DIN + kt + c8, &Alds[buf][row * LDA + c8]);
      async_cp16(WtS + (size_t)row  * DIN + kt + c8, &Blds[buf][row * LDA + c8]);
    }
  };

  issue(0, 0);
  for (int it = 0; it < DIN / KT; ++it) {
    const int cur = it & 1;
    if (it > 0) __syncthreads();                 // buf cur^1 free to overwrite
    if (it + 1 < DIN / KT) {
      issue(cur ^ 1, (it + 1) * KT);
      wait_asynccnt<4>();                        // drain previous batch (in order)
    } else {
      wait_asynccnt<0>();
    }
    __syncthreads();                             // all waves' tile landed
#else
  for (int it = 0; it < DIN / KT; ++it) {
    const int cur = 0;
    const int kt = it * KT;
    if (it > 0) __syncthreads();
#pragma unroll
    for (int j = 0; j < 2; ++j) {
      int lin = tid + j * 256;
      int row = lin >> 2;
      int c8  = (lin & 3) << 3;
      int grow = gm0 + row;
      if (grow >= N_NODES) grow = N_NODES - 1;
      *(uint4*)&Alds[0][row * LDA + c8] = *(const uint4*)(xd  + (size_t)grow * DIN + kt + c8);
      *(uint4*)&Blds[0][row * LDA + c8] = *(const uint4*)(WtS + (size_t)row  * DIN + kt + c8);
    }
    __syncthreads();
#endif

    // A fragment: lane<16 -> K {0..7,16..23} of row m; lane>=16 -> {8..15,24..31}
    const int arow = wave * 16 + m16;
    const int akb  = hi * 8;
    bf16x8 alo = *(const bf16x8*)&Alds[cur][arow * LDA + akb];
    bf16x8 ahi = *(const bf16x8*)&Alds[cur][arow * LDA + akb + 16];
    bf16x16 afrag;
#pragma unroll
    for (int i = 0; i < 8; ++i) { afrag[i] = alo[i]; afrag[8 + i] = ahi[i]; }

#pragma unroll
    for (int nt = 0; nt < 8; ++nt) {
      const int bcol = nt * 16 + m16;            // lane owns one column
      const int bks  = hi * 8;                   // K {0..7,16..23} / {8..15,24..31}
      bf16x8 b0 = *(const bf16x8*)&Blds[cur][bcol * LDA + bks];
      bf16x8 b1 = *(const bf16x8*)&Blds[cur][bcol * LDA + bks + 16];
      bf16x16 bfrag;
#pragma unroll
      for (int i = 0; i < 8; ++i) { bfrag[i] = b0[i]; bfrag[8 + i] = b1[i]; }
      acc[nt] = __builtin_amdgcn_wmma_f32_16x16x32_bf16(
          false, afrag, false, bfrag, (short)0, acc[nt], false, false);
    }
  }

  // C layout: VGPR r -> row gm0 + wave*16 + r + 8*hi, col = nt*16 + m16.
  unsigned short* preS = pre + (size_t)s * N_NODES * DOUT;
#pragma unroll
  for (int nt = 0; nt < 8; ++nt) {
    int col = nt * 16 + m16;
#pragma unroll
    for (int r = 0; r < 8; ++r) {
      int row = gm0 + wave * 16 + r + hi * 8;
      if (row < N_NODES)
        preS[(size_t)row * DOUT + col] = f32_to_bf16(acc[nt][r]);
    }
  }
}

// --------------------------------------------------------------------------
// Kernel 3: COO SpMM scatter. Wave per edge (scalarized descriptors),
// bf16 gather from L2-resident table, native fp32 global atomics.
// --------------------------------------------------------------------------
__global__ __launch_bounds__(256)
void spmm_atomic(const int* __restrict__ rows, const int* __restrict__ cols,
                 const float* __restrict__ vals,
                 const unsigned short* __restrict__ pre,
                 float* __restrict__ out) {
  const int s    = blockIdx.y;
  const int lane = threadIdx.x & 31;
  // wave-uniform edge index -> scalar (SMEM) descriptor loads
  const int e = __builtin_amdgcn_readfirstlane((int)blockIdx.x * 8 + (int)(threadIdx.x >> 5));
  const size_t eb = (size_t)s * NEDGE + e;
  const int   r = rows[eb];
  const int   c = cols[eb];
  const float v = vals[eb];
  const ushort4 h =
      *(const ushort4*)(pre + (size_t)s * N_NODES * DOUT + (size_t)c * DOUT + lane * 4);
  float* o = out + (size_t)s * N_NODES * DOUT + (size_t)r * DOUT + lane * 4;
  unsafeAtomicAdd(o + 0, v * bf16_to_f32(h.x));
  unsafeAtomicAdd(o + 1, v * bf16_to_f32(h.y));
  unsafeAtomicAdd(o + 2, v * bf16_to_f32(h.z));
  unsafeAtomicAdd(o + 3, v * bf16_to_f32(h.w));
}

// --------------------------------------------------------------------------
// Kernel 4: in-place ReLU (float4).
// --------------------------------------------------------------------------
__global__ __launch_bounds__(256)
void relu_inplace(float* __restrict__ out, int n4) {
  int i = blockIdx.x * 256 + threadIdx.x;
  if (i >= n4) return;
  float4 v = ((float4*)out)[i];
  v.x = fmaxf(v.x, 0.0f); v.y = fmaxf(v.y, 0.0f);
  v.z = fmaxf(v.z, 0.0f); v.w = fmaxf(v.w, 0.0f);
  ((float4*)out)[i] = v;
}

// --------------------------------------------------------------------------
extern "C" void kernel_launch(void* const* d_in, const int* in_sizes, int n_in,
                              void* d_out, int out_size, void* d_ws, size_t ws_size,
                              hipStream_t stream) {
  const float* x    = (const float*)d_in[0];
  const float* W    = (const float*)d_in[1];
  const int*   rows = (const int*)d_in[2];
  const int*   cols = (const int*)d_in[3];
  const float* vals = (const float*)d_in[4];
  float*       out  = (float*)d_out;

  // Workspace layout (~77.1 MB):
  //   [0, 51.2M)        xd  : dropout(x) bf16
  //   [51.2M, 76.8M)    pre : pre_sup bf16
  //   [76.8M, 77.06M)   Wt  : weights bf16, transposed [S][N][K]
  unsigned short* xd  = (unsigned short*)d_ws;
  unsigned short* pre = (unsigned short*)((char*)d_ws + (size_t)N_NODES * DIN * 2);
  unsigned short* Wt  = (unsigned short*)((char*)d_ws + (size_t)N_NODES * DIN * 2
                                                      + (size_t)NSUP * N_NODES * DOUT * 2);

  dropout_cast<<<12500, 256, 0, stream>>>(x, xd);
  wt_cast<<<512, 256, 0, stream>>>(W, Wt);

  gemm_wmma<<<dim3((N_NODES + BM - 1) / BM, NSUP), 256, 0, stream>>>(xd, Wt, pre);

  (void)hipMemsetAsync(d_out, 0, (size_t)NSUP * N_NODES * DOUT * sizeof(float), stream);

  spmm_atomic<<<dim3(NEDGE / 8, NSUP), 256, 0, stream>>>(rows, cols, vals, pre, out);

  relu_inplace<<<(NSUP * N_NODES * DOUT / 4 + 255) / 256, 256, 0, stream>>>(
      out, NSUP * N_NODES * DOUT / 4);
}
